// TransformerBlock_67525475828367
// MI455X (gfx1250) — matrix operand order repaired and checked by reading, not compiled, not run on hardware
//
#include <hip/hip_runtime.h>

// ---------------------------------------------------------------------------
// Point-Transformer block for MI455X (gfx1250, wave32, WMMA).
// B=4, N=4096, K=16, D_POINTS=64, D_MODEL=128.
//
// Roofline: ~45 GFLOP dense; fusing the per-point [16 x 128] neighbor block
// through 5 back-to-back v_wmma_f32_16x16x32_f16 GEMMs keeps all [B,N,K,D]
// intermediates in accumulator registers (the 16x16 f32 C/D layout puts the
// K-neighbor axis on M-rows -> softmax over K is register math + one
// shfl_xor(16)). Only HBM intermediates: knn idx (1MB), f16 f (4MB), q (8MB).
// ---------------------------------------------------------------------------

typedef _Float16 h8  __attribute__((ext_vector_type(8)));
typedef _Float16 v16h __attribute__((ext_vector_type(16)));
typedef float    v8f  __attribute__((ext_vector_type(8)));

__device__ __forceinline__ v8f zero8() {
  v8f z;
#pragma unroll
  for (int i = 0; i < 8; ++i) z[i] = 0.0f;
  return z;
}

__device__ __forceinline__ v16h pack16(h8 lo, h8 hi) {
  v16h a;
#pragma unroll
  for (int i = 0; i < 8; ++i) { a[i] = lo[i]; a[i + 8] = hi[i]; }
  return a;
}

// A fragment (16xK tile, 16-bit): lane = M (mod 16); halves 0-7 = K[kb..kb+7],
// halves 8-15 = K[kb+16..kb+23], kb = 32c + 8*(lane>=16). Row-major buf, ld halves.
__device__ __forceinline__ v16h load_A(const _Float16* buf, int ld, int lane, int c) {
  int M  = lane & 15;
  int kb = c * 32 + ((lane & 16) >> 1);
  const _Float16* p = buf + M * ld + kb;
  return pack16(*(const h8*)p, *(const h8*)(p + 16));
}

// B fragment from pre-swizzled weights: 16 contiguous halves per (tile t, chunk c, lane).
__device__ __forceinline__ v16h load_B(const _Float16* Wsw, int lane, int t, int c, int C) {
  const _Float16* p = Wsw + ((((t * C + c) << 5) + lane) << 4);
  return pack16(*(const h8*)p, *(const h8*)(p + 8));
}

__device__ __forceinline__ v8f wmma16(v16h a, v16h b, v8f c) {
  return __builtin_amdgcn_wmma_f32_16x16x32_f16(false, a, false, b, (short)0, c,
                                                false, false);
}

// ---------------------------------------------------------------------------
// k0: swizzle f32 weight [Krows x Ncols] into f16 B-operand order.
// element (k,col): c=k/32, hi=((k%32)/8)&1, h=k%8+8*((k%32)/16), lane=(col%16)+16*hi, t=col/16
// ---------------------------------------------------------------------------
__global__ void k0_swizzle(const float* __restrict__ src, _Float16* __restrict__ dst,
                           int Krows, int Ncols) {
  int i = blockIdx.x * 256 + threadIdx.x;
  if (i >= Krows * Ncols) return;
  int k = i / Ncols, col = i % Ncols;
  int C = Krows >> 5;
  int c = k >> 5, r2 = k & 31, hi = (r2 >> 3) & 1, h = (r2 & 7) + ((r2 >> 4) << 3);
  int lane = (col & 15) + (hi << 4);
  int t = col >> 4;
  dst[((((t * C + c) << 5) + lane) << 4) + h] = (_Float16)src[i];
}

// ---------------------------------------------------------------------------
// k1: brute-force KNN, 16 smallest of d2 = |q|^2+|p|^2-2 q.p (ref formula),
// ties keep lower index (matches top_k of -d2). One thread per query.
// ---------------------------------------------------------------------------
__global__ void __launch_bounds__(256) k1_knn(const float* __restrict__ xyz,
                                              int* __restrict__ knnIdx) {
  __shared__ float sx[256], sy[256], sz[256], sq[256];
  int tid  = threadIdx.x;
  int qi   = blockIdx.x * 256 + tid;      // global point, same batch per block
  int base = (qi >> 12) << 12;
  float qx = xyz[qi * 3 + 0], qy = xyz[qi * 3 + 1], qz = xyz[qi * 3 + 2];
  float qsq = qx * qx + qy * qy + qz * qz;
  float bd[16]; int bi[16];
#pragma unroll
  for (int t = 0; t < 16; ++t) { bd[t] = 3.4e38f; bi[t] = 0; }
  for (int j0 = 0; j0 < 4096; j0 += 256) {
    int pt = base + j0 + tid;
    float x = xyz[pt * 3 + 0], y = xyz[pt * 3 + 1], z = xyz[pt * 3 + 2];
    sx[tid] = x; sy[tid] = y; sz[tid] = z; sq[tid] = x * x + y * y + z * z;
    __syncthreads();
    for (int jj = 0; jj < 256; ++jj) {
      float d = qsq + sq[jj] - 2.0f * (qx * sx[jj] + qy * sy[jj] + qz * sz[jj]);
      int idx = j0 + jj;
      if (d < bd[15]) {
#pragma unroll
        for (int t = 0; t < 16; ++t) {
          if (d < bd[t]) {
            float td = bd[t]; int ti = bi[t];
            bd[t] = d; bi[t] = idx; d = td; idx = ti;
          }
        }
      }
    }
    __syncthreads();
  }
#pragma unroll
  for (int t = 0; t < 16; ++t) knnIdx[qi * 16 + t] = bi[t];
}

// ---------------------------------------------------------------------------
// k2: f = features@Wf1 + bf1 (store f16), q = f@Wq (store f32).
// One wave per 16-row M-tile; 4 waves/block; no barriers (per-wave LDS slices,
// DS ops are in-order within a wave).
// ---------------------------------------------------------------------------
__global__ void __launch_bounds__(128, 1)
k2_fc1_q(const float* __restrict__ feat, const float* __restrict__ bf1,
         const _Float16* __restrict__ Wf1sw, const _Float16* __restrict__ Wqsw,
         _Float16* __restrict__ fOut, float* __restrict__ qOut) {
  __shared__ _Float16 aBuf[4][16 * 64];
  __shared__ _Float16 fBufS[4][16 * 128];
  int lane = threadIdx.x & 31, wv = threadIdx.x >> 5;
  int r0 = (blockIdx.x * 4 + wv) * 16;

  // stage features rows (f32 -> f16) into LDS A tile [16][64]
  for (int i = lane; i < 16 * 64; i += 32)
    aBuf[wv][i] = (_Float16)feat[(r0 + (i >> 6)) * 64 + (i & 63)];

  const _Float16* aw = &aBuf[wv][0];
#pragma unroll
  for (int t = 0; t < 8; ++t) {
    v8f acc = zero8();
#pragma unroll
    for (int c = 0; c < 2; ++c)
      acc = wmma16(load_A(aw, 64, lane, c), load_B(Wf1sw, lane, t, c, 2), acc);
    int col = t * 16 + (lane & 15);
    float bb = bf1[col];
#pragma unroll
    for (int r = 0; r < 8; ++r) {
      int row = r + ((lane & 16) >> 1);
      _Float16 h = (_Float16)(acc[r] + bb);
      fBufS[wv][row * 128 + col] = h;
      fOut[(size_t)(r0 + row) * 128 + col] = h;
    }
  }

  const _Float16* fw = &fBufS[wv][0];
#pragma unroll
  for (int t = 0; t < 8; ++t) {
    v8f acc = zero8();
#pragma unroll
    for (int c = 0; c < 4; ++c)
      acc = wmma16(load_A(fw, 128, lane, c), load_B(Wqsw, lane, t, c, 4), acc);
    int col = t * 16 + (lane & 15);
#pragma unroll
    for (int r = 0; r < 8; ++r) {
      int row = r + ((lane & 16) >> 1);
      qOut[(size_t)(r0 + row) * 128 + col] = acc[r];
    }
  }
}

// ---------------------------------------------------------------------------
// k3: fused attention core. One wave per query point; 4 waves/block.
// pos_enc / v+pos / attn logits live entirely in WMMA accumulators
// (reg r -> neighbor row M = r + 8*(lane>=16); column N = 16t + lane%16).
// ---------------------------------------------------------------------------
__global__ void __launch_bounds__(128, 1)
k3_attn(const float* __restrict__ xyz, const float* __restrict__ feat,
        const float* __restrict__ Wd1, const float* __restrict__ bd1,
        const float* __restrict__ bd2, const float* __restrict__ bg1,
        const float* __restrict__ bg2, const float* __restrict__ Wo,
        const float* __restrict__ bo, const int* __restrict__ knnIdx,
        const _Float16* __restrict__ fBuf, const float* __restrict__ qBuf,
        const _Float16* __restrict__ Wd2sw, const _Float16* __restrict__ Wksw,
        const _Float16* __restrict__ Wvsw, const _Float16* __restrict__ Wg1sw,
        const _Float16* __restrict__ Wg2sw, float* __restrict__ out) {
  __shared__ _Float16 aBuf[4][16 * 128];
  __shared__ int   sIdx[4][16];
  __shared__ float sRes[4][128];

  int lane = threadIdx.x & 31, wv = threadIdx.x >> 5;
  int p = blockIdx.x * 4 + wv;          // global point id
  int base = (p >> 12) << 12;           // batch base
  const _Float16* aw = &aBuf[wv][0];
  int col0 = lane & 15;
  int rowh = (lane & 16) >> 1;          // 0 or 8

  if (lane < 16) sIdx[wv][lane] = knnIdx[p * 16 + lane];

  // --- hidden1 = relu(rel @ Wd1 + bd1) -> aBuf (f16) -------------------------
  {
    int k  = lane & 15;
    int dh = (lane & 16) << 2;          // 0 or 64
    int nb = base + sIdx[wv][k];
    float rx = xyz[p * 3 + 0] - xyz[nb * 3 + 0];
    float ry = xyz[p * 3 + 1] - xyz[nb * 3 + 1];
    float rz = xyz[p * 3 + 2] - xyz[nb * 3 + 2];
    for (int d = 0; d < 64; ++d) {
      int dd = dh + d;
      float h = rx * Wd1[dd] + ry * Wd1[128 + dd] + rz * Wd1[256 + dd] + bd1[dd];
      aBuf[wv][k * 128 + dd] = (_Float16)(h > 0.0f ? h : 0.0f);
    }
  }

  // --- pos_enc = hidden1 @ Wd2 + bd2 (kept in regs) --------------------------
  v8f pos[8];
#pragma unroll
  for (int t = 0; t < 8; ++t) {
    v8f acc = zero8();
#pragma unroll
    for (int c = 0; c < 4; ++c)
      acc = wmma16(load_A(aw, 128, lane, c), load_B(Wd2sw, lane, t, c, 4), acc);
    float bb = bd2[t * 16 + col0];
#pragma unroll
    for (int r = 0; r < 8; ++r) pos[t][r] = acc[r] + bb;
  }

  // --- gather knn_f rows (f16) into aBuf -------------------------------------
  {
    int k = lane & 15;
    int hb = (lane & 16) << 2;          // 0 or 64 halves
    const _Float16* src = fBuf + (size_t)(base + sIdx[wv][k]) * 128 + hb;
    _Float16* dst = &aBuf[wv][k * 128 + hb];
#pragma unroll
    for (int i = 0; i < 64; i += 8) *(h8*)(dst + i) = *(const h8*)(src + i);
  }

  // --- vp = knn_f @ Wv + pos (regs) ------------------------------------------
  v8f vp[8];
#pragma unroll
  for (int t = 0; t < 8; ++t) {
    v8f acc = zero8();
#pragma unroll
    for (int c = 0; c < 4; ++c)
      acc = wmma16(load_A(aw, 128, lane, c), load_B(Wvsw, lane, t, c, 4), acc);
#pragma unroll
    for (int r = 0; r < 8; ++r) vp[t][r] = acc[r] + pos[t][r];
  }

  // --- att = q - knn_f @ Wk + pos (regs) -------------------------------------
  v8f att[8];
#pragma unroll
  for (int t = 0; t < 8; ++t) {
    v8f acc = zero8();
#pragma unroll
    for (int c = 0; c < 4; ++c)
      acc = wmma16(load_A(aw, 128, lane, c), load_B(Wksw, lane, t, c, 4), acc);
    float qv = qBuf[(size_t)p * 128 + t * 16 + col0];
#pragma unroll
    for (int r = 0; r < 8; ++r) att[t][r] = qv - acc[r] + pos[t][r];
  }

  // write attn input to aBuf (all Wk/Wv reads of aBuf are done; DS in-order)
#pragma unroll
  for (int t = 0; t < 8; ++t) {
    int col = t * 16 + col0;
#pragma unroll
    for (int r = 0; r < 8; ++r)
      aBuf[wv][(r + rowh) * 128 + col] = (_Float16)att[t][r];
  }

  // --- h2 = relu(att @ Wg1 + bg1) --------------------------------------------
#pragma unroll
  for (int t = 0; t < 8; ++t) {
    v8f acc = zero8();
#pragma unroll
    for (int c = 0; c < 4; ++c)
      acc = wmma16(load_A(aw, 128, lane, c), load_B(Wg1sw, lane, t, c, 4), acc);
    float bb = bg1[t * 16 + col0];
#pragma unroll
    for (int r = 0; r < 8; ++r) {
      float h = acc[r] + bb;
      att[t][r] = h > 0.0f ? h : 0.0f;
    }
  }
#pragma unroll
  for (int t = 0; t < 8; ++t) {
    int col = t * 16 + col0;
#pragma unroll
    for (int r = 0; r < 8; ++r)
      aBuf[wv][(r + rowh) * 128 + col] = (_Float16)att[t][r];
  }

  // --- logits = h2 @ Wg2 + bg2 -----------------------------------------------
#pragma unroll
  for (int t = 0; t < 8; ++t) {
    v8f acc = zero8();
#pragma unroll
    for (int c = 0; c < 4; ++c)
      acc = wmma16(load_A(aw, 128, lane, c), load_B(Wg2sw, lane, t, c, 4), acc);
    float bb = bg2[t * 16 + col0];
#pragma unroll
    for (int r = 0; r < 8; ++r) att[t][r] = acc[r] + bb;
  }

  // --- softmax over K (M rows) per channel + sum attn*(v+pos) ----------------
  const float inv_s = 0.08838834764831845f;  // 1/sqrt(128)
#pragma unroll
  for (int t = 0; t < 8; ++t) {
    float m = -3.4e38f;
#pragma unroll
    for (int r = 0; r < 8; ++r) m = fmaxf(m, att[t][r] * inv_s);
    m = fmaxf(m, __shfl_xor(m, 16, 32));   // combine M=0..7 with M=8..15
    float e[8], s = 0.0f;
#pragma unroll
    for (int r = 0; r < 8; ++r) { e[r] = __expf(att[t][r] * inv_s - m); s += e[r]; }
    s += __shfl_xor(s, 16, 32);
    float part = 0.0f;
#pragma unroll
    for (int r = 0; r < 8; ++r) part += (e[r] / s) * vp[t][r];
    part += __shfl_xor(part, 16, 32);
    if (lane < 16) sRes[wv][t * 16 + lane] = part;
  }

  // --- out = res @ Wo + bo + features ----------------------------------------
#pragma unroll
  for (int cc = 0; cc < 2; ++cc) {
    int c = lane * 2 + cc;
    float acc = bo[c] + feat[(size_t)p * 64 + c];
    for (int d = 0; d < 128; ++d) acc += sRes[wv][d] * Wo[d * 64 + c];
    out[(size_t)p * 64 + c] = acc;
  }
}

// ---------------------------------------------------------------------------
extern "C" void kernel_launch(void* const* d_in, const int* in_sizes, int n_in,
                              void* d_out, int out_size, void* d_ws, size_t ws_size,
                              hipStream_t stream) {
  (void)in_sizes; (void)n_in; (void)out_size; (void)ws_size;
  const float* xyz  = (const float*)d_in[0];
  const float* feat = (const float*)d_in[1];
  const float* Wd1  = (const float*)d_in[2];
  const float* bd1  = (const float*)d_in[3];
  const float* Wd2  = (const float*)d_in[4];
  const float* bd2  = (const float*)d_in[5];
  const float* Wf1  = (const float*)d_in[6];
  const float* bf1  = (const float*)d_in[7];
  const float* Wq   = (const float*)d_in[8];
  const float* Wk   = (const float*)d_in[9];
  const float* Wv   = (const float*)d_in[10];
  const float* Wg1  = (const float*)d_in[11];
  const float* bg1  = (const float*)d_in[12];
  const float* Wg2  = (const float*)d_in[13];
  const float* bg2  = (const float*)d_in[14];
  const float* Wo   = (const float*)d_in[15];
  const float* bo   = (const float*)d_in[16];
  float* out = (float*)d_out;

  char* ws = (char*)d_ws;
  int*      knnIdx = (int*)ws;       ws += (size_t)16384 * 16 * 4;
  _Float16* fBuf   = (_Float16*)ws;  ws += (size_t)16384 * 128 * 2;
  float*    qBuf   = (float*)ws;     ws += (size_t)16384 * 128 * 4;
  _Float16* Wf1sw  = (_Float16*)ws;  ws += (size_t)64 * 128 * 2;
  _Float16* Wqsw   = (_Float16*)ws;  ws += (size_t)128 * 128 * 2;
  _Float16* Wd2sw  = (_Float16*)ws;  ws += (size_t)128 * 128 * 2;
  _Float16* Wksw   = (_Float16*)ws;  ws += (size_t)128 * 128 * 2;
  _Float16* Wvsw   = (_Float16*)ws;  ws += (size_t)128 * 128 * 2;
  _Float16* Wg1sw  = (_Float16*)ws;  ws += (size_t)128 * 128 * 2;
  _Float16* Wg2sw  = (_Float16*)ws;  ws += (size_t)128 * 128 * 2;

  k0_swizzle<<<32, 256, 0, stream>>>(Wf1, Wf1sw, 64, 128);
  k0_swizzle<<<64, 256, 0, stream>>>(Wq,  Wqsw,  128, 128);
  k0_swizzle<<<64, 256, 0, stream>>>(Wd2, Wd2sw, 128, 128);
  k0_swizzle<<<64, 256, 0, stream>>>(Wk,  Wksw,  128, 128);
  k0_swizzle<<<64, 256, 0, stream>>>(Wv,  Wvsw,  128, 128);
  k0_swizzle<<<64, 256, 0, stream>>>(Wg1, Wg1sw, 128, 128);
  k0_swizzle<<<64, 256, 0, stream>>>(Wg2, Wg2sw, 128, 128);

  k1_knn<<<64, 256, 0, stream>>>(xyz, knnIdx);
  k2_fc1_q<<<256, 128, 0, stream>>>(feat, bf1, Wf1sw, Wqsw, fBuf, qBuf);
  k3_attn<<<4096, 128, 0, stream>>>(xyz, feat, Wd1, bd1, bd2, bg1, bg2, Wo, bo,
                                    knnIdx, fBuf, qBuf, Wd2sw, Wksw, Wvsw,
                                    Wg1sw, Wg2sw, out);
}
